// EEHGCN_82085414961677
// MI455X (gfx1250) — matrix operand order repaired and checked by reading, not compiled, not run on hardware
//
#include <hip/hip_runtime.h>
#include <hip/hip_bf16.h>

typedef __attribute__((ext_vector_type(16))) _Float16 v16h;
typedef __attribute__((ext_vector_type(8)))  float    v8f;

#define DIM 64
#define K3  192   // 3*DIM

// ---------------------------------------------------------------------------
// SpMM: y[row[e], :] += val[e] * x[col[e], :]   (COO, atomic scatter)
// 16 lanes per nnz, 4 floats per lane (float4 gather, 4 atomic f32 adds).
// ---------------------------------------------------------------------------
__global__ __launch_bounds__(256)
void spmm_kernel(const int* __restrict__ row, const int* __restrict__ col,
                 const float* __restrict__ val, const float* __restrict__ x,
                 float* __restrict__ y, int nnz) {
    long long t = (long long)blockIdx.x * blockDim.x + threadIdx.x;
    int e  = (int)(t >> 4);
    int d4 = ((int)t & 15) * 4;
    if (e >= nnz) return;
    int c = col[e];
    int r = row[e];
    float v = val[e];
    float4 xv = *(const float4*)(x + (size_t)c * DIM + d4);
    float* yp = y + (size_t)r * DIM + d4;
    atomicAdd(yp + 0, v * xv.x);
    atomicAdd(yp + 1, v * xv.y);
    atomicAdd(yp + 2, v * xv.z);
    atomicAdd(yp + 3, v * xv.w);
}

// ---------------------------------------------------------------------------
// GEMM + leaky_relu:  yout = lrelu( [ego | nbin | nbout] @ W ),  W: [192][64]
// 4 wave32 per block, each wave owns a 16-row tile.
// W is converted to f16 and transposed into LDS as wT[n][k] so a B fragment
// (K contiguous per lane) is one aligned 32-byte LDS read.
// ---------------------------------------------------------------------------
__global__ __launch_bounds__(128)
void gemm_lrelu_kernel(const float* __restrict__ ego,
                       const float* __restrict__ nbin,
                       const float* __restrict__ nbout,
                       const float* __restrict__ W,
                       float* __restrict__ yout,
                       int nrows) {
    __shared__ __align__(32) _Float16 wT[DIM * K3];   // [n][k], 24 KB

    // Cooperative load + f16 convert + transpose of the 192x64 weight.
    for (int idx = threadIdx.x; idx < K3 * DIM; idx += 128) {
        int k = idx >> 6, n = idx & 63;
        wT[n * K3 + k] = (_Float16)W[idx];
    }
    __syncthreads();

    int wave = threadIdx.x >> 5;
    int lane = threadIdx.x & 31;
    int m    = lane & 15;
    int hi   = lane >> 4;

    int rowbase = (blockIdx.x * 4 + wave) * 16;
    int rowm    = rowbase + m;
    int rowc    = rowm < nrows ? rowm : (nrows - 1);   // clamp for loads only

    const float* srcs[3] = { ego   + (size_t)rowc * DIM,
                             nbin  + (size_t)rowc * DIM,
                             nbout + (size_t)rowc * DIM };

    // A fragments (contiguous K order, matching B):
    //   lanes 0-15  : halves 0-15 -> K=k0+0..15
    //   lanes 16-31 : halves 0-15 -> K=k0+16..31
    v16h a[6];
#pragma unroll
    for (int kt = 0; kt < 6; ++kt) {
        const float* s = srcs[kt >> 1];
        int kb = (kt & 1) * 32;
        int ka = kb + hi * 16;         // halves 0..7  (8 contiguous f32)
        int kc = kb + hi * 16 + 8;     // halves 8..15 (8 contiguous f32)
        float4 x0 = *(const float4*)(s + ka);
        float4 x1 = *(const float4*)(s + ka + 4);
        float4 x2 = *(const float4*)(s + kc);
        float4 x3 = *(const float4*)(s + kc + 4);
        v16h t;
        t[0]  = (_Float16)x0.x; t[1]  = (_Float16)x0.y;
        t[2]  = (_Float16)x0.z; t[3]  = (_Float16)x0.w;
        t[4]  = (_Float16)x1.x; t[5]  = (_Float16)x1.y;
        t[6]  = (_Float16)x1.z; t[7]  = (_Float16)x1.w;
        t[8]  = (_Float16)x2.x; t[9]  = (_Float16)x2.y;
        t[10] = (_Float16)x2.z; t[11] = (_Float16)x2.w;
        t[12] = (_Float16)x3.x; t[13] = (_Float16)x3.y;
        t[14] = (_Float16)x3.z; t[15] = (_Float16)x3.w;
        a[kt] = t;
    }

    v8f acc[4];
#pragma unroll
    for (int nt = 0; nt < 4; ++nt)
#pragma unroll
        for (int r = 0; r < 8; ++r) acc[nt][r] = 0.0f;

    // B in contiguous K order (same as A): lane -> N = nt*16 + (lane&15),
    // halves h -> K = k0 + (lane>>4)*16 + h  (contiguous in wT[n][k]).
#pragma unroll
    for (int nt = 0; nt < 4; ++nt) {
        const _Float16* bp = &wT[(nt * 16 + m) * K3 + hi * 16];
#pragma unroll
        for (int kt = 0; kt < 6; ++kt) {
            v16h b = *(const v16h*)(bp + kt * 32);
            acc[nt] = __builtin_amdgcn_wmma_f32_16x16x32_f16(
                false, a[kt], false, b, (short)0, acc[nt], false, false);
        }
    }

    // Epilogue: D layout — VGPR r, lane l holds [M = r + 8*(l>>4)][N = l&15].
    int orow0 = rowbase + hi * 8;
#pragma unroll
    for (int nt = 0; nt < 4; ++nt) {
        int ocol = nt * 16 + m;
#pragma unroll
        for (int r = 0; r < 8; ++r) {
            float v = acc[nt][r];
            v = v > 0.0f ? v : 0.01f * v;          // leaky_relu(0.01)
            int orow = orow0 + r;
            if (orow < nrows) yout[(size_t)orow * DIM + ocol] = v;
        }
    }
}

// ---------------------------------------------------------------------------
// Per-row L2 norm (with eps clamp, matching F.normalize eps=1e-12)
// ---------------------------------------------------------------------------
__global__ __launch_bounds__(256)
void norm_kernel(const float* __restrict__ x, float* __restrict__ norms, int n) {
    int i = blockIdx.x * blockDim.x + threadIdx.x;
    if (i >= n) return;
    const float4* p = (const float4*)(x + (size_t)i * DIM);
    float s = 0.0f;
#pragma unroll
    for (int j = 0; j < DIM / 4; ++j) {
        float4 v = p[j];
        s += v.x * v.x + v.y * v.y + v.z * v.z + v.w * v.w;
    }
    float nn = sqrtf(s);
    norms[i] = nn > 1e-12f ? nn : 1e-12f;
}

// ---------------------------------------------------------------------------
// Per-relation segment-sum of (optionally normalized) rows into out[:, colbase:+64]
// LDS [16][64] f32 accumulator per block, then 1024 global atomics per block.
// ---------------------------------------------------------------------------
__global__ __launch_bounds__(256)
void relsum_kernel(const float* __restrict__ x, const int* __restrict__ attr,
                   const float* __restrict__ norms, int use_norm,
                   float* __restrict__ out, int colbase, int n) {
    __shared__ float acc[16 * DIM];
    for (int i = threadIdx.x; i < 16 * DIM; i += 256) acc[i] = 0.0f;
    __syncthreads();

    int d  = threadIdx.x & 63;
    int rl = threadIdx.x >> 6;            // 0..3
    int base = blockIdx.x * 1024;
    int end  = base + 1024 < n ? base + 1024 : n;
    for (int r = base + rl; r < end; r += 4) {
        float v = x[(size_t)r * DIM + d];
        if (use_norm) v /= norms[r];
        atomicAdd(&acc[attr[r] * DIM + d], v);
    }
    __syncthreads();

    for (int i = threadIdx.x; i < 16 * DIM; i += 256) {
        int rel = i >> 6, dd = i & 63;
        atomicAdd(&out[rel * K3 + colbase + dd], acc[i]);
    }
}

// ---------------------------------------------------------------------------
// Launch
// ---------------------------------------------------------------------------
extern "C" void kernel_launch(void* const* d_in, const int* in_sizes, int n_in,
                              void* d_out, int out_size, void* d_ws, size_t ws_size,
                              hipStream_t stream) {
    const float* edge_emb = (const float*)d_in[0];
    const float* w0       = (const float*)d_in[1];
    const float* w1       = (const float*)d_in[2];
    const int*   in_row   = (const int*)d_in[3];
    const int*   in_col   = (const int*)d_in[4];
    const float* in_val   = (const float*)d_in[5];
    const int*   out_row  = (const int*)d_in[6];
    const int*   out_col  = (const int*)d_in[7];
    const float* out_val  = (const float*)d_in[8];
    const int*   attr     = (const int*)d_in[9];

    int N   = in_sizes[0] / DIM;   // 100000
    int nnz = in_sizes[3];         // 1600000
    float* out = (float*)d_out;

    size_t rowsz = (size_t)N * DIM;
    float* ego1   = (float*)d_ws;
    float* ego2   = ego1  + rowsz;
    float* nb_in  = ego2  + rowsz;
    float* nb_out = nb_in + rowsz;
    float* norms  = nb_out + rowsz;

    int rsBlocks = (N + 1023) / 1024;
    int spBlocks = (int)(((size_t)nnz * 16 + 255) / 256);
    int gmBlocks = (N + 63) / 64;
    int nmBlocks = (N + 255) / 256;

    hipMemsetAsync(out, 0, (size_t)out_size * sizeof(float), stream);

    // all_emb[0] contribution: raw edge_emb, columns [0,64)
    relsum_kernel<<<rsBlocks, 256, 0, stream>>>(edge_emb, attr, norms, 0, out, 0, N);

    // ---- Layer 1 ----
    hipMemsetAsync(nb_in,  0, rowsz * sizeof(float), stream);
    hipMemsetAsync(nb_out, 0, rowsz * sizeof(float), stream);
    spmm_kernel<<<spBlocks, 256, 0, stream>>>(in_row,  in_col,  in_val,  edge_emb, nb_in,  nnz);
    spmm_kernel<<<spBlocks, 256, 0, stream>>>(out_row, out_col, out_val, edge_emb, nb_out, nnz);
    gemm_lrelu_kernel<<<gmBlocks, 128, 0, stream>>>(edge_emb, nb_in, nb_out, w0, ego1, N);
    norm_kernel<<<nmBlocks, 256, 0, stream>>>(ego1, norms, N);
    relsum_kernel<<<rsBlocks, 256, 0, stream>>>(ego1, attr, norms, 1, out, 64, N);

    // ---- Layer 2 (input is UNnormalized ego1, matching the reference) ----
    hipMemsetAsync(nb_in,  0, rowsz * sizeof(float), stream);
    hipMemsetAsync(nb_out, 0, rowsz * sizeof(float), stream);
    spmm_kernel<<<spBlocks, 256, 0, stream>>>(in_row,  in_col,  in_val,  ego1, nb_in,  nnz);
    spmm_kernel<<<spBlocks, 256, 0, stream>>>(out_row, out_col, out_val, ego1, nb_out, nnz);
    gemm_lrelu_kernel<<<gmBlocks, 128, 0, stream>>>(ego1, nb_in, nb_out, w1, ego2, N);
    norm_kernel<<<nmBlocks, 256, 0, stream>>>(ego2, norms, N);
    relsum_kernel<<<rsBlocks, 256, 0, stream>>>(ego2, attr, norms, 1, out, 128, N);
}
